// AGSRNet_18854906430032
// MI455X (gfx1250) — matrix operand order repaired and checked by reading, not compiled, not run on hardware
//
#include <hip/hip_runtime.h>

// ---------------------------------------------------------------------------
// Types for WMMA (gfx1250, wave32)
// ---------------------------------------------------------------------------
typedef __attribute__((ext_vector_type(16))) __bf16          v16bf;
typedef __attribute__((ext_vector_type(8)))  unsigned int    v8u;
typedef __attribute__((ext_vector_type(8)))  float           v8f;

static __device__ __forceinline__ unsigned short f2bf(float f) {
    unsigned int u = __float_as_uint(f);
    unsigned int r = (u + 0x7FFFu + ((u >> 16) & 1u)) >> 16;   // RNE
    return (unsigned short)r;
}

// ---------------------------------------------------------------------------
// bf16-WMMA GEMM: C = A @ B (+bias / relu / abs), row-major f32 in/out.
// Block = 256 threads (8 waves), 64x64 C tile; each wave computes 16x32 of C
// (two accumulators sharing one A fragment -> 2 WMMA per A-fragment load).
// LDS: A staged row-major (lane fragment = 2x ds_load_b128), B staged
// TRANSPOSED (Bt[n][k]) so a B fragment is also 2x ds_load_b128.
// flags: bit0 = +bias[col], bit1 = relu, bit2 = abs
// ---------------------------------------------------------------------------
__global__ __launch_bounds__(256)
void gemm_wmma_bf16(const float* __restrict__ Ag, const float* __restrict__ Bg,
                    const float* __restrict__ bias, float* __restrict__ Cg,
                    int M, int N, int K, int flags)
{
    __shared__ alignas(16) unsigned short As[64][40];   // 64 rows (M) x 32 (K), stride 40
    __shared__ alignas(16) unsigned short Bt[64][40];   // 64 rows (N) x 32 (K), stride 40

    const int tid  = threadIdx.x;
    const int lane = tid & 31;
    const int wave = tid >> 5;
    const int r16  = lane & 15;
    const int hi   = lane >> 4;
    const int waveM = (wave >> 1) << 4;   // 0,16,32,48
    const int waveN = (wave & 1) << 5;    // 0,32
    const int rowBase = blockIdx.y * 64;
    const int colBase = blockIdx.x * 64;

    v8f acc0 = {}, acc1 = {};

    for (int k0 = 0; k0 < K; k0 += 32) {
        // ---- stage A tile: 64(M) x 32(K) ----
        #pragma unroll
        for (int it = 0; it < 2; ++it) {
            int i4 = it * 256 + tid;          // 512 float4 slots
            int rr = i4 >> 3;                 // row 0..63
            int c4 = (i4 & 7) << 2;           // col 0,4,..,28
            int am = rowBase + rr, ak = k0 + c4;
            unsigned short h0, h1, h2, h3;
            if (am < M && ak + 3 < K) {
                const float4 f = *(const float4*)(Ag + (size_t)am * K + ak);
                h0 = f2bf(f.x); h1 = f2bf(f.y); h2 = f2bf(f.z); h3 = f2bf(f.w);
            } else {
                float x0 = (am < M && ak + 0 < K) ? Ag[(size_t)am * K + ak + 0] : 0.f;
                float x1 = (am < M && ak + 1 < K) ? Ag[(size_t)am * K + ak + 1] : 0.f;
                float x2 = (am < M && ak + 2 < K) ? Ag[(size_t)am * K + ak + 2] : 0.f;
                float x3 = (am < M && ak + 3 < K) ? Ag[(size_t)am * K + ak + 3] : 0.f;
                h0 = f2bf(x0); h1 = f2bf(x1); h2 = f2bf(x2); h3 = f2bf(x3);
            }
            unsigned long long pk = (unsigned long long)h0
                                  | ((unsigned long long)h1 << 16)
                                  | ((unsigned long long)h2 << 32)
                                  | ((unsigned long long)h3 << 48);
            *(unsigned long long*)&As[rr][c4] = pk;     // ds_store_b64
        }
        // ---- stage B tile transposed: Bt[n][k], source 32(K) x 64(N) ----
        #pragma unroll
        for (int it = 0; it < 2; ++it) {
            int i4 = it * 256 + tid;          // 512 float4 slots
            int rr = i4 >> 4;                 // k row 0..31
            int c4 = (i4 & 15) << 2;          // n col 0,4,..,60
            int bk = k0 + rr, bn = colBase + c4;
            float x0, x1, x2, x3;
            if (bk < K && bn + 3 < N) {
                const float4 f = *(const float4*)(Bg + (size_t)bk * N + bn);
                x0 = f.x; x1 = f.y; x2 = f.z; x3 = f.w;
            } else {
                x0 = (bk < K && bn + 0 < N) ? Bg[(size_t)bk * N + bn + 0] : 0.f;
                x1 = (bk < K && bn + 1 < N) ? Bg[(size_t)bk * N + bn + 1] : 0.f;
                x2 = (bk < K && bn + 2 < N) ? Bg[(size_t)bk * N + bn + 2] : 0.f;
                x3 = (bk < K && bn + 3 < N) ? Bg[(size_t)bk * N + bn + 3] : 0.f;
            }
            Bt[c4 + 0][rr] = f2bf(x0);
            Bt[c4 + 1][rr] = f2bf(x1);
            Bt[c4 + 2][rr] = f2bf(x2);
            Bt[c4 + 3][rr] = f2bf(x3);
        }
        __syncthreads();

        // ---- fragments (2x ds_load_b128 each) ----
        const int m = waveM + r16;
        const uint4 al = *(const uint4*)&As[m][hi * 8];        // K = hi*8 + 0..7
        const uint4 ah = *(const uint4*)&As[m][16 + hi * 8];   // K = 16 + hi*8 + 0..7
        v8u av; av[0] = al.x; av[1] = al.y; av[2] = al.z; av[3] = al.w;
                av[4] = ah.x; av[5] = ah.y; av[6] = ah.z; av[7] = ah.w;
        const v16bf a = __builtin_bit_cast(v16bf, av);

        const int n0 = waveN + r16;
        const uint4 b0l = *(const uint4*)&Bt[n0][hi * 8];      // K = hi*8 + 0..7
        const uint4 b0h = *(const uint4*)&Bt[n0][16 + hi * 8]; // K = 16 + hi*8 + 0..7
        v8u bv0; bv0[0] = b0l.x; bv0[1] = b0l.y; bv0[2] = b0l.z; bv0[3] = b0l.w;
                 bv0[4] = b0h.x; bv0[5] = b0h.y; bv0[6] = b0h.z; bv0[7] = b0h.w;
        const v16bf b0 = __builtin_bit_cast(v16bf, bv0);

        const int n1 = n0 + 16;
        const uint4 b1l = *(const uint4*)&Bt[n1][hi * 8];
        const uint4 b1h = *(const uint4*)&Bt[n1][16 + hi * 8];
        v8u bv1; bv1[0] = b1l.x; bv1[1] = b1l.y; bv1[2] = b1l.z; bv1[3] = b1l.w;
                 bv1[4] = b1h.x; bv1[5] = b1h.y; bv1[6] = b1h.z; bv1[7] = b1h.w;
        const v16bf b1 = __builtin_bit_cast(v16bf, bv1);

        acc0 = __builtin_amdgcn_wmma_f32_16x16x32_bf16(
                   false, a, false, b0, (short)0, acc0, false, false);
        acc1 = __builtin_amdgcn_wmma_f32_16x16x32_bf16(
                   false, a, false, b1, (short)0, acc1, false, false);
        __syncthreads();
    }

    const int col0 = colBase + waveN + r16;
    const int col1 = col0 + 16;
    #pragma unroll
    for (int v = 0; v < 8; ++v) {
        int row = rowBase + waveM + v + hi * 8;
        if (row < M) {
            if (col0 < N) {
                float val = acc0[v];
                if (flags & 1) val += bias[col0];
                if (flags & 2) val = fmaxf(val, 0.f);
                if (flags & 4) val = fabsf(val);
                Cg[(size_t)row * N + col0] = val;
            }
            if (col1 < N) {
                float val = acc1[v];
                if (flags & 1) val += bias[col1];
                if (flags & 2) val = fmaxf(val, 0.f);
                if (flags & 4) val = fabsf(val);
                Cg[(size_t)row * N + col1] = val;
            }
        }
    }
}

// ---------------------------------------------------------------------------
// Small support kernels
// ---------------------------------------------------------------------------
__global__ void row_rsqrt(const float* __restrict__ lr, float* __restrict__ r, int n)
{
    __shared__ float red[256];
    int row = blockIdx.x;
    float s = 0.f;
    for (int j = threadIdx.x; j < n; j += 256) s += lr[(size_t)row * n + j];
    red[threadIdx.x] = s;
    __syncthreads();
    for (int off = 128; off > 0; off >>= 1) {
        if (threadIdx.x < off) red[threadIdx.x] += red[threadIdx.x + off];
        __syncthreads();
    }
    if (threadIdx.x == 0) {
        float sum = red[0];
        r[row] = (sum > 0.f) ? rsqrtf(sum) : 0.f;
    }
}

__global__ void build_adj(const float* __restrict__ lr, const float* __restrict__ r,
                          float* __restrict__ A, int n)
{
    size_t idx = (size_t)blockIdx.x * blockDim.x + threadIdx.x;
    if (idx >= (size_t)n * n) return;
    int i = (int)(idx / n), j = (int)(idx % n);
    A[idx] = lr[(size_t)j * n + i] * r[i] * r[j];   // (lr * r).T * r
}

__global__ void pool_scores(const float* __restrict__ X, const float* __restrict__ pw,
                            const float* __restrict__ pb, float* __restrict__ sc, int n)
{
    int i = blockIdx.x * blockDim.x + threadIdx.x;
    if (i >= n) return;
    float d = 0.f;
    for (int c = 0; c < 320; ++c) d += X[(size_t)i * 320 + c] * pw[c];
    float v = (d + pb[0]) * 0.01f;
    sc[i] = 1.f / (1.f + __expf(-v));
}

// full bitonic sort of up to 1024 (key, index) pairs; desc=1 -> descending key,
// ties broken by ascending index (matches jax.lax.top_k ordering).
__global__ __launch_bounds__(1024)
void bitonic_sort1024(const float* __restrict__ keys, int n, int desc,
                      float* __restrict__ okeys, int* __restrict__ oidx)
{
    __shared__ float sk[1024];
    __shared__ int   si[1024];
    int t = threadIdx.x;
    if (t < n) { sk[t] = keys[t]; si[t] = t; }
    else       { sk[t] = desc ? -__builtin_inff() : __builtin_inff(); si[t] = 0x7fffffff; }
    __syncthreads();
    for (int k = 2; k <= 1024; k <<= 1) {
        for (int j = k >> 1; j > 0; j >>= 1) {
            int ixj = t ^ j;
            if (ixj > t) {
                float ka = sk[t], kb = sk[ixj];
                int   ia = si[t], ib = si[ixj];
                bool bBefore = desc ? ((kb > ka) || (kb == ka && ib < ia))
                                    : ((kb < ka) || (kb == ka && ib < ia));
                bool dir = ((t & k) == 0);
                if (dir ? bBefore : !bBefore) {
                    sk[t] = kb; sk[ixj] = ka; si[t] = ib; si[ixj] = ia;
                }
            }
            __syncthreads();
        }
    }
    if (t < n) { okeys[t] = sk[t]; oidx[t] = si[t]; }
}

__global__ void gather_rows_scale(float* __restrict__ out, const float* __restrict__ in,
                                  const int* __restrict__ idx, const float* __restrict__ vals,
                                  int k)
{
    size_t t = (size_t)blockIdx.x * blockDim.x + threadIdx.x;
    if (t >= (size_t)k * 320) return;
    int r = (int)(t / 320), c = (int)(t % 320);
    out[t] = in[(size_t)idx[r] * 320 + c] * vals[r];
}

__global__ void gather_adj(float* __restrict__ out, const float* __restrict__ in,
                           const int* __restrict__ idx, int k, int nold)
{
    size_t t = (size_t)blockIdx.x * blockDim.x + threadIdx.x;
    if (t >= (size_t)k * k) return;
    int i = (int)(t / k), j = (int)(t % k);
    out[t] = in[(size_t)idx[i] * nold + idx[j]];
}

__global__ void zero_f(float* __restrict__ x, size_t n)
{
    size_t t = (size_t)blockIdx.x * blockDim.x + threadIdx.x;
    if (t < n) x[t] = 0.f;
}

__global__ void scatter_rows(float* __restrict__ out, const float* __restrict__ in,
                             const int* __restrict__ idx, int k)
{
    size_t t = (size_t)blockIdx.x * blockDim.x + threadIdx.x;
    if (t >= (size_t)k * 320) return;
    int r = (int)(t / 320), c = (int)(t % 320);
    out[(size_t)idx[r] * 320 + c] = in[t];
}

__global__ void add_inplace(float* __restrict__ x, const float* __restrict__ y, size_t n)
{
    size_t t = (size_t)blockIdx.x * blockDim.x + threadIdx.x;
    if (t < n) x[t] += y[t];
}

__global__ void concat2(float* __restrict__ out, const float* __restrict__ x,
                        const float* __restrict__ org, int n)
{
    size_t t = (size_t)blockIdx.x * blockDim.x + threadIdx.x;
    if (t >= (size_t)n * 640) return;
    int r = (int)(t / 640), c = (int)(t % 640);
    out[t] = (c < 320) ? x[(size_t)r * 320 + c] : org[(size_t)r * 320 + (c - 320)];
}

__global__ void transpose_f(float* __restrict__ out, const float* __restrict__ in,
                            int R, int C)
{
    size_t t = (size_t)blockIdx.x * blockDim.x + threadIdx.x;
    if (t >= (size_t)R * C) return;
    int r = (int)(t / C), c = (int)(t % C);
    out[(size_t)c * R + r] = in[t];
}

__global__ void gsr_half_add(float* __restrict__ a, const float* __restrict__ g)
{
    size_t t = (size_t)blockIdx.x * blockDim.x + threadIdx.x;
    if (t >= (size_t)2048 * 1024) return;
    int i = (int)(t / 1024), j = (int)(t % 1024);
    a[t] = g[(size_t)i * 2048 + j] + g[(size_t)i * 2048 + 1024 + j];
}

__global__ void set_diag(float* __restrict__ M, int n, float v)
{
    int i = blockIdx.x * blockDim.x + threadIdx.x;
    if (i < n) M[(size_t)i * n + i] = v;
}

// dst = |(src + src^T)/2| with unit diagonal (works in-place: src == dst)
__global__ void sym_abs_diag(const float* __restrict__ src, float* __restrict__ dst, int n)
{
    size_t t = (size_t)blockIdx.x * blockDim.x + threadIdx.x;
    if (t >= (size_t)n * n) return;
    int i = (int)(t / n), j = (int)(t % n);
    if (i > j) return;
    if (i == j) { dst[t] = 1.f; return; }
    float m = fabsf(0.5f * (src[(size_t)i * n + j] + src[(size_t)j * n + i]));
    dst[(size_t)i * n + j] = m;
    dst[(size_t)j * n + i] = m;
}

__global__ void set_identity(float* __restrict__ U, int n)
{
    size_t t = (size_t)blockIdx.x * blockDim.x + threadIdx.x;
    if (t >= (size_t)n * n) return;
    U[t] = ((t / n) == (t % n)) ? 1.f : 0.f;
}

// ---------------------------------------------------------------------------
// Cyclic Jacobi eigensolver kernels (round-robin pairing, 512 pairs / round)
// ---------------------------------------------------------------------------
__global__ void jacobi_pairs(const float* __restrict__ Aw, int n, int round,
                             float* __restrict__ cA, float* __restrict__ sA,
                             int* __restrict__ pA, int* __restrict__ qA)
{
    int i = threadIdx.x;                 // 512 threads
    int m = n - 1;
    int p, q;
    if (i == 0) { p = m; q = round % m; }
    else {
        p = (round + i) % m;
        q = (round - i + 2 * m) % m;
    }
    if (p > q) { int tt = p; p = q; q = tt; }
    float app = Aw[(size_t)p * n + p];
    float aqq = Aw[(size_t)q * n + q];
    float apq = Aw[(size_t)p * n + q];
    float c = 1.f, s = 0.f;
    if (fabsf(apq) > 1e-12f) {
        float tau = (aqq - app) / (2.f * apq);
        float tt  = (tau >= 0.f ? 1.f : -1.f) / (fabsf(tau) + sqrtf(1.f + tau * tau));
        c = rsqrtf(1.f + tt * tt);
        s = tt * c;
    }
    cA[i] = c; sA[i] = s; pA[i] = p; qA[i] = q;
}

__global__ void jacobi_rows(float* __restrict__ Aw, int n,
                            const float* __restrict__ cA, const float* __restrict__ sA,
                            const int* __restrict__ pA, const int* __restrict__ qA)
{
    int pair = blockIdx.y;
    int j = blockIdx.x * blockDim.x + threadIdx.x;
    if (j >= n) return;
    int p = pA[pair], q = qA[pair];
    float c = cA[pair], s = sA[pair];
    float ap = Aw[(size_t)p * n + j], aq = Aw[(size_t)q * n + j];
    Aw[(size_t)p * n + j] = c * ap - s * aq;
    Aw[(size_t)q * n + j] = s * ap + c * aq;
}

__global__ void jacobi_cols(float* __restrict__ Aw, float* __restrict__ U, int n,
                            const float* __restrict__ cA, const float* __restrict__ sA,
                            const int* __restrict__ pA, const int* __restrict__ qA)
{
    int pair = blockIdx.y;
    int i = blockIdx.x * blockDim.x + threadIdx.x;
    if (i >= n) return;
    int p = pA[pair], q = qA[pair];
    float c = cA[pair], s = sA[pair];
    float ap = Aw[(size_t)i * n + p], aq = Aw[(size_t)i * n + q];
    Aw[(size_t)i * n + p] = c * ap - s * aq;
    Aw[(size_t)i * n + q] = s * ap + c * aq;
    float up = U[(size_t)i * n + p], uq = U[(size_t)i * n + q];
    U[(size_t)i * n + p] = c * up - s * uq;
    U[(size_t)i * n + q] = s * up + c * uq;
}

__global__ void diag_extract(const float* __restrict__ Aw, float* __restrict__ ev, int n)
{
    int i = blockIdx.x * blockDim.x + threadIdx.x;
    if (i < n) ev[i] = Aw[(size_t)i * n + i];
}

// UT[k][i] = U[i][perm[k]]  (fused column-gather + transpose)
__global__ void gather_UT(float* __restrict__ UT, const float* __restrict__ U,
                          const int* __restrict__ perm, int n)
{
    size_t t = (size_t)blockIdx.x * blockDim.x + threadIdx.x;
    if (t >= (size_t)n * n) return;
    int k = (int)(t / n), i = (int)(t % n);
    UT[t] = U[(size_t)i * n + perm[k]];
}

// ---------------------------------------------------------------------------
// Workspace layout (floats)
// ---------------------------------------------------------------------------
static constexpr size_t LRN = 1024, HRN = 2048, DIM = 320;
static constexpr size_t F_R    = 0;
static constexpr size_t F_A    = F_R    + 1024;
static constexpr size_t F_AW   = F_A    + LRN * LRN;
static constexpr size_t F_U    = F_AW   + LRN * LRN;
static constexpr size_t F_UT   = F_U    + LRN * LRN;
static constexpr size_t F_EV   = F_UT   + LRN * LRN;
static constexpr size_t F_EVS  = F_EV   + 1024;
static constexpr size_t F_PERM = F_EVS  + 1024;       // int
static constexpr size_t F_C    = F_PERM + 1024;
static constexpr size_t F_S    = F_C    + 512;
static constexpr size_t F_P    = F_S    + 512;        // int
static constexpr size_t F_Q    = F_P    + 512;        // int
static constexpr size_t F_IDX  = F_Q    + 512;        // int, 4 levels x 1024
static constexpr size_t F_VALS = F_IDX  + 4096;
static constexpr size_t F_SC   = F_VALS + 4096;
static constexpr size_t F_DX   = F_SC   + 1024;       // 4 x (1024*320)
static constexpr size_t F_AL   = F_DX   + 4 * LRN * DIM;   // 4 x 1024^2
static constexpr size_t F_XB1  = F_AL   + 4 * LRN * LRN;
static constexpr size_t F_XB2  = F_XB1  + LRN * DIM;
static constexpr size_t F_XU   = F_XB2  + LRN * DIM;
static constexpr size_t F_XCAT = F_XU   + LRN * DIM;
static constexpr size_t F_AMAT = F_XCAT + LRN * 2 * DIM;
static constexpr size_t F_UTN  = F_AMAT + HRN * LRN;
static constexpr size_t F_Z    = F_UTN  + LRN * HRN;
static constexpr size_t F_H1   = F_Z    + HRN * HRN;
static constexpr size_t F_TA   = F_H1   + HRN * LRN;
static constexpr size_t F_TB   = F_TA   + HRN * HRN;

extern "C" void kernel_launch(void* const* d_in, const int* in_sizes, int n_in,
                              void* d_out, int out_size, void* d_ws, size_t ws_size,
                              hipStream_t stream)
{
    (void)in_sizes; (void)n_in; (void)out_size; (void)ws_size;

    const float* lr       = (const float*)d_in[0];
    const float* gsr_w    = (const float*)d_in[1];
    const float* start_w  = (const float*)d_in[2];
    const float* start_b  = (const float*)d_in[3];
    const float* down_w   = (const float*)d_in[4];
    const float* down_b   = (const float*)d_in[5];
    const float* pool_w   = (const float*)d_in[6];
    const float* pool_b   = (const float*)d_in[7];
    const float* bottom_w = (const float*)d_in[8];
    const float* bottom_b = (const float*)d_in[9];
    const float* end_w    = (const float*)d_in[10];
    const float* end_b    = (const float*)d_in[11];
    const float* up_w     = (const float*)d_in[12];
    const float* up_b     = (const float*)d_in[13];
    const float* gc1_w    = (const float*)d_in[14];
    const float* gc2_w    = (const float*)d_in[15];

    float* ws   = (float*)d_ws;
    float* outp = (float*)d_out;

    // d_out layout: z | net_outs | start_outs | adj
    float* zOut   = outp;
    float* netOut = outp + HRN * HRN;
    float* startX = netOut + LRN * HRN;          // start_outs == org_X
    float* adjOut = startX + LRN * DIM;

    float* rV   = ws + F_R;
    float* A    = ws + F_A;
    float* Aw   = ws + F_AW;
    float* U    = ws + F_U;
    float* UT   = ws + F_UT;
    float* ev   = ws + F_EV;
    float* evs  = ws + F_EVS;
    int*   perm = (int*)(ws + F_PERM);
    float* cArr = ws + F_C;
    float* sArr = ws + F_S;
    int*   pArr = (int*)(ws + F_P);
    int*   qArr = (int*)(ws + F_Q);
    int*   idxB = (int*)(ws + F_IDX);
    float* valB = ws + F_VALS;
    float* sc   = ws + F_SC;
    float* dX   = ws + F_DX;                     // stride LRN*DIM per level
    float* Alev = ws + F_AL;                     // stride LRN*LRN per level
    float* Xb1  = ws + F_XB1;
    float* Xb2  = ws + F_XB2;
    float* Xu   = ws + F_XU;
    float* Xcat = ws + F_XCAT;
    float* aMat = ws + F_AMAT;
    float* UtN  = ws + F_UTN;
    float* Zb   = ws + F_Z;
    float* h1   = ws + F_H1;
    float* tA   = ws + F_TA;
    float* tB   = ws + F_TB;

    auto g1 = [](size_t n) { return dim3((unsigned)((n + 255) / 256)); };
    auto gemm = [&](float* C, const float* Ao, const float* Bo, const float* bias,
                    int M, int N, int K, int flags) {
        dim3 grid((N + 63) / 64, (M + 63) / 64);
        gemm_wmma_bf16<<<grid, dim3(256), 0, stream>>>(Ao, Bo, bias, C, M, N, K, flags);
    };

    const int n0 = (int)LRN;
    const int ns[5] = {1024, 921, 644, 386, 193};

    // ---- normalized adjacency ----
    row_rsqrt<<<dim3(n0), dim3(256), 0, stream>>>(lr, rV, n0);
    build_adj<<<g1((size_t)n0 * n0), dim3(256), 0, stream>>>(lr, rV, A, n0);

    // ---- start GCN (X = I  =>  A @ start_w + b) ----
    gemm(startX, A, start_w, start_b, n0, (int)DIM, n0, 1);

    // ---- down path ----
    const float* Xcur = startX;
    for (int lvl = 0; lvl < 4; ++lvl) {
        int ncur = ns[lvl], knext = ns[lvl + 1];
        const float* Acur = (lvl == 0) ? A : (Alev + (size_t)(lvl - 1) * LRN * LRN);
        float* dXl = dX + (size_t)lvl * LRN * DIM;
        gemm(tA, Acur, Xcur, nullptr, ncur, (int)DIM, ncur, 0);
        gemm(dXl, tA, down_w + (size_t)lvl * DIM * DIM, down_b + (size_t)lvl * DIM,
             ncur, (int)DIM, (int)DIM, 1);
        pool_scores<<<g1(ncur), dim3(256), 0, stream>>>(dXl, pool_w + (size_t)lvl * DIM,
                                                        pool_b + lvl, sc, ncur);
        bitonic_sort1024<<<dim3(1), dim3(1024), 0, stream>>>(sc, ncur, 1,
                                                             valB + lvl * 1024, idxB + lvl * 1024);
        float* Xn = (lvl & 1) ? Xb2 : Xb1;
        gather_rows_scale<<<g1((size_t)knext * DIM), dim3(256), 0, stream>>>(
            Xn, dXl, idxB + lvl * 1024, valB + lvl * 1024, knext);
        gather_adj<<<g1((size_t)knext * knext), dim3(256), 0, stream>>>(
            Alev + (size_t)lvl * LRN * LRN, Acur, idxB + lvl * 1024, knext, ncur);
        Xcur = Xn;
    }

    // ---- bottom GCN (193 nodes) ----
    gemm(tA, Alev + 3 * LRN * LRN, Xcur, nullptr, ns[4], (int)DIM, ns[4], 0);
    gemm(Xb1, tA, bottom_w, bottom_b, ns[4], (int)DIM, (int)DIM, 1);
    Xcur = Xb1;

    // ---- up path ----
    for (int i = 0; i < 4; ++i) {
        int j = 3 - i;
        int nj = ns[j], kc = ns[j + 1];
        const float* Aj = (j == 0) ? A : (Alev + (size_t)(j - 1) * LRN * LRN);
        zero_f<<<g1((size_t)nj * DIM), dim3(256), 0, stream>>>(Xu, (size_t)nj * DIM);
        scatter_rows<<<g1((size_t)kc * DIM), dim3(256), 0, stream>>>(Xu, Xcur, idxB + j * 1024, kc);
        gemm(tA, Aj, Xu, nullptr, nj, (int)DIM, nj, 0);
        float* Xn = (Xcur == Xb1) ? Xb2 : Xb1;
        gemm(Xn, tA, up_w + (size_t)i * DIM * DIM, up_b + (size_t)i * DIM,
             nj, (int)DIM, (int)DIM, 1);
        add_inplace<<<g1((size_t)nj * DIM), dim3(256), 0, stream>>>(
            Xn, dX + (size_t)j * LRN * DIM, (size_t)nj * DIM);
        Xcur = Xn;
    }

    // ---- end GCN: net_outs = (A @ [X, org_X]) @ end_w + end_b ----
    concat2<<<g1((size_t)n0 * 640), dim3(256), 0, stream>>>(Xcat, Xcur, startX, n0);
    gemm(tA, A, Xcat, nullptr, n0, 640, n0, 0);
    gemm(netOut, tA, end_w, end_b, n0, (int)HRN, 640, 1);

    // ---- eigh(A) via cyclic Jacobi ----
    hipMemcpyAsync(Aw, A, (size_t)n0 * n0 * sizeof(float), hipMemcpyDeviceToDevice, stream);
    set_identity<<<g1((size_t)n0 * n0), dim3(256), 0, stream>>>(U, n0);
    const int SWEEPS = 8;
    for (int sw = 0; sw < SWEEPS; ++sw) {
        for (int rd = 0; rd < n0 - 1; ++rd) {
            jacobi_pairs<<<dim3(1), dim3(512), 0, stream>>>(Aw, n0, rd, cArr, sArr, pArr, qArr);
            jacobi_rows<<<dim3(4, 512), dim3(256), 0, stream>>>(Aw, n0, cArr, sArr, pArr, qArr);
            jacobi_cols<<<dim3(4, 512), dim3(256), 0, stream>>>(Aw, U, n0, cArr, sArr, pArr, qArr);
        }
    }
    diag_extract<<<g1(n0), dim3(256), 0, stream>>>(Aw, ev, n0);
    bitonic_sort1024<<<dim3(1), dim3(1024), 0, stream>>>(ev, n0, 0, evs, perm);  // ascending
    gather_UT<<<g1((size_t)n0 * n0), dim3(256), 0, stream>>>(UT, U, perm, n0);

    // ---- GSR: adj = |a @ U^T @ net_outs| with unit diag ----
    gsr_half_add<<<g1(HRN * LRN), dim3(256), 0, stream>>>(aMat, gsr_w);
    gemm(UtN, UT, netOut, nullptr, n0, (int)HRN, n0, 0);
    gemm(adjOut, aMat, UtN, nullptr, (int)HRN, (int)HRN, n0, 4 /*abs*/);
    set_diag<<<g1(HRN), dim3(256), 0, stream>>>(adjOut, (int)HRN, 1.f);

    // ---- Z = sym/abs/diag(adj @ adj^T) ----
    transpose_f<<<g1(HRN * HRN), dim3(256), 0, stream>>>(tA, adjOut, (int)HRN, (int)HRN);
    gemm(Zb, adjOut, tA, nullptr, (int)HRN, (int)HRN, (int)HRN, 0);
    sym_abs_diag<<<g1(HRN * HRN), dim3(256), 0, stream>>>(Zb, Zb, (int)HRN);

    // ---- refinement GCN layers ----
    gemm(tA, Zb, gc1_w, nullptr, (int)HRN, (int)LRN, (int)HRN, 0);
    gemm(h1, adjOut, tA, nullptr, (int)HRN, (int)LRN, (int)HRN, 2 /*relu*/);
    gemm(tA, h1, gc2_w, nullptr, (int)HRN, (int)HRN, (int)LRN, 0);
    gemm(tB, adjOut, tA, nullptr, (int)HRN, (int)HRN, (int)HRN, 2 /*relu*/);

    // ---- z = sym/abs/diag(h2) ----
    sym_abs_diag<<<g1(HRN * HRN), dim3(256), 0, stream>>>(tB, zOut, (int)HRN);
}
